// CriticNetwork_38611755991585
// MI455X (gfx1250) — matrix operand order repaired and checked by reading, not compiled, not run on hardware
//
#include <hip/hip_runtime.h>
#include <hip/hip_bf16.h>

typedef __attribute__((ext_vector_type(16))) _Float16 v16h;
typedef __attribute__((ext_vector_type(8)))  float    v8f;

#define NPB   100000          // nodes per batch
#define NROWS (2 * NPB)       // B*N flattened rows
#define FIN   128
#define HID   16
#define NCOL  1000
#define CFEAT 64

// ---------------------------------------------------------------------------
// ws layout (floats): [0]=node_sum_b0 [1]=node_sum_b1 [2]=col_sum_b0 [3]=col_sum_b1
// ---------------------------------------------------------------------------
__global__ __launch_bounds__(32) void zero_ws_kernel(float* __restrict__ ws) {
  if (threadIdx.x < 4) ws[threadIdx.x] = 0.0f;
}

// ---------------------------------------------------------------------------
// Fused node path: relu(relu(x@W1+b1)@W2+b2)@Wn+bn, summed per batch.
// Layer 1 (K=128) via v_wmma_f32_16x16x32_f16; layers 2/3 (K=16) via VALU.
// One wave handles 32 consecutive rows per iteration (two 16-row WMMA tiles).
// ---------------------------------------------------------------------------
__global__ __launch_bounds__(256) void node_path_kernel(
    const float* __restrict__ x,
    const float* __restrict__ W1, const float* __restrict__ b1,
    const float* __restrict__ W2, const float* __restrict__ b2,
    const float* __restrict__ Wn, const float* __restrict__ bn,
    float* __restrict__ ws)
{
  __shared__ float sX1[8][2][16 * 16];   // per-wave x1 tiles (8 waves x 2 tiles)

  const int lane = threadIdx.x & 31;
  const int wav  = threadIdx.x >> 5;
  const int g    = lane >> 4;            // half-wave group
  const int mr   = lane & 15;            // A row / B,C column index

  // W1 B-fragments for the four K=32 chunks (loop-invariant -> registers).
  // Dense 16-bit B 32x16 layout matching A: lane group g holds col N=lane%16,
  // K = 8g + e for e < 8, K = 16 + 8g + (e - 8) for e >= 8.
  v16h bw[4];
#pragma unroll
  for (int ch = 0; ch < 4; ++ch) {
#pragma unroll
    for (int e = 0; e < 16; ++e)
      bw[ch][e] = (_Float16)W1[(ch * 32 + (e < 8 ? g * 8 + e : 16 + g * 8 + (e - 8))) * HID + mr];
  }
  const float b1v = b1[mr];
  const float bnv = bn[0];

  float accB0 = 0.0f, accB1 = 0.0f;

  const int wavesTotal = gridDim.x * (blockDim.x >> 5);
  const int wid        = blockIdx.x * (blockDim.x >> 5) + wav;
  const int npairs     = NROWS / 32;     // 6250; 32 | 100000 -> no batch straddle

  for (int p = wid; p < npairs; p += wavesTotal) {
    const int row0 = p * 32;

    // -------- Layer 1: two 16x128 tiles x W1 via WMMA --------
#pragma unroll
    for (int tt = 0; tt < 2; ++tt) {
      const float* rp = x + (size_t)(row0 + tt * 16 + mr) * FIN;
      v8f acc = {};
#pragma unroll
      for (int ch = 0; ch < 4; ++ch) {
        // Documented f16 A 16x32 layout: group g holds row M=lane%16,
        // K in {8g..8g+7} U {16+8g..16+8g+7}, elements 2v+h sequential.
        const float* cp = rp + ch * 32 + g * 8;
        float4 q0 = *(const float4*)(cp);
        float4 q1 = *(const float4*)(cp + 4);
        float4 q2 = *(const float4*)(cp + 16);
        float4 q3 = *(const float4*)(cp + 20);
        v16h a;
        a[0]  = (_Float16)q0.x; a[1]  = (_Float16)q0.y;
        a[2]  = (_Float16)q0.z; a[3]  = (_Float16)q0.w;
        a[4]  = (_Float16)q1.x; a[5]  = (_Float16)q1.y;
        a[6]  = (_Float16)q1.z; a[7]  = (_Float16)q1.w;
        a[8]  = (_Float16)q2.x; a[9]  = (_Float16)q2.y;
        a[10] = (_Float16)q2.z; a[11] = (_Float16)q2.w;
        a[12] = (_Float16)q3.x; a[13] = (_Float16)q3.y;
        a[14] = (_Float16)q3.z; a[15] = (_Float16)q3.w;
        acc = __builtin_amdgcn_wmma_f32_16x16x32_f16(
                  false, a, false, bw[ch], (short)0, acc, false, false);
      }
      // bias + relu; park tile in per-wave LDS (C layout: col=mr, rows j+8g)
#pragma unroll
      for (int j = 0; j < 8; ++j) {
        float v = acc[j] + b1v;
        sX1[wav][tt][(j + 8 * g) * 16 + mr] = v > 0.0f ? v : 0.0f;
      }
    }
    asm volatile("s_wait_dscnt 0" ::: "memory");

    // -------- Layers 2+3: each lane owns one of the 32 rows --------
    const float4* xr = (const float4*)&sX1[wav][g][mr * 16];
    float4 x0 = xr[0], x1 = xr[1], x2 = xr[2], x3 = xr[3];
    float xv[16] = {x0.x, x0.y, x0.z, x0.w, x1.x, x1.y, x1.z, x1.w,
                    x2.x, x2.y, x2.z, x2.w, x3.x, x3.y, x3.z, x3.w};
    float y[16];
#pragma unroll
    for (int n = 0; n < 16; ++n) y[n] = b2[n];          // uniform -> s_loads
#pragma unroll
    for (int k = 0; k < 16; ++k) {
      const float xk = xv[k];
#pragma unroll
      for (int n = 0; n < 16; ++n) y[n] += xk * W2[k * HID + n];
    }
    float s = bnv;
#pragma unroll
    for (int n = 0; n < 16; ++n) {
      float t = y[n] > 0.0f ? y[n] : 0.0f;
      s += t * Wn[n];
    }
    if (row0 < NPB) accB0 += s; else accB1 += s;
  }

  // wave32 tree reduction, then one atomic per wave per batch
#pragma unroll
  for (int off = 16; off > 0; off >>= 1) {
    accB0 += __shfl_down(accB0, off, 32);
    accB1 += __shfl_down(accB1, off, 32);
  }
  if (lane == 0) {
    atomicAdd(&ws[0], accB0);
    atomicAdd(&ws[1], accB1);
  }
}

// ---------------------------------------------------------------------------
// Col path: relu(cf@Wc1+bc1)@Wc2+bc2, summed per batch. 2000 rows -> VALU.
// ---------------------------------------------------------------------------
__global__ __launch_bounds__(256) void col_path_kernel(
    const float* __restrict__ cf,
    const float* __restrict__ Wc1, const float* __restrict__ bc1,
    const float* __restrict__ Wc2, const float* __restrict__ bc2,
    float* __restrict__ ws)
{
  const float bc2v = bc2[0];
  for (int r = blockIdx.x * blockDim.x + threadIdx.x; r < 2 * NCOL;
       r += gridDim.x * blockDim.x) {
    const float* cp = cf + (size_t)r * CFEAT;
    float y[16];
#pragma unroll
    for (int n = 0; n < 16; ++n) y[n] = bc1[n];
    for (int k = 0; k < CFEAT; ++k) {
      const float ck = cp[k];
#pragma unroll
      for (int n = 0; n < 16; ++n) y[n] += ck * Wc1[k * HID + n];
    }
    float s = bc2v;
#pragma unroll
    for (int n = 0; n < 16; ++n) {
      float t = y[n] > 0.0f ? y[n] : 0.0f;
      s += t * Wc2[n];
    }
    atomicAdd(&ws[2 + (r >= NCOL ? 1 : 0)], s);
  }
}

// ---------------------------------------------------------------------------
// Head: means -> [na, ca] @ Wf (+bf, relu) @ Wo + bo  -> out[B]
// ---------------------------------------------------------------------------
__global__ __launch_bounds__(32) void head_kernel(
    const float* __restrict__ ws,
    const float* __restrict__ Wf, const float* __restrict__ bf,
    const float* __restrict__ Wo, const float* __restrict__ bo,
    float* __restrict__ out)
{
  const int b = threadIdx.x;
  if (b < 2) {
    const float na = ws[b]     * (1.0f / (float)NPB);
    const float ca = ws[2 + b] * (1.0f / (float)NCOL);
    float r = bo[0];
#pragma unroll
    for (int n = 0; n < 16; ++n) {
      float h = na * Wf[n] + ca * Wf[16 + n] + bf[n];
      h = h > 0.0f ? h : 0.0f;
      r += h * Wo[n];
    }
    out[b] = r;
  }
}

// ---------------------------------------------------------------------------
extern "C" void kernel_launch(void* const* d_in, const int* in_sizes, int n_in,
                              void* d_out, int out_size, void* d_ws, size_t ws_size,
                              hipStream_t stream) {
  (void)in_sizes; (void)n_in; (void)out_size; (void)ws_size;
  const float* x   = (const float*)d_in[0];
  const float* cf  = (const float*)d_in[1];
  // d_in[2] = edge_index (int64): provably unused. The torch-faithful
  // expand+reshape makes row==col, so GCN normalization + scatter is the
  // identity and each conv is exactly x@W + b.
  const float* W1  = (const float*)d_in[3];
  const float* b1  = (const float*)d_in[4];
  const float* W2  = (const float*)d_in[5];
  const float* b2  = (const float*)d_in[6];
  const float* Wn  = (const float*)d_in[7];
  const float* bn  = (const float*)d_in[8];
  const float* Wc1 = (const float*)d_in[9];
  const float* bc1 = (const float*)d_in[10];
  const float* Wc2 = (const float*)d_in[11];
  const float* bc2 = (const float*)d_in[12];
  const float* Wf  = (const float*)d_in[13];
  const float* bfp = (const float*)d_in[14];
  const float* Wo  = (const float*)d_in[15];
  const float* bo  = (const float*)d_in[16];
  float* out = (float*)d_out;
  float* ws  = (float*)d_ws;

  hipLaunchKernelGGL(zero_ws_kernel,  dim3(1),   dim3(32),  0, stream, ws);
  hipLaunchKernelGGL(node_path_kernel, dim3(160), dim3(256), 0, stream,
                     x, W1, b1, W2, b2, Wn, bn, ws);
  hipLaunchKernelGGL(col_path_kernel, dim3(8),   dim3(256), 0, stream,
                     cf, Wc1, bc1, Wc2, bc2, ws);
  hipLaunchKernelGGL(head_kernel,     dim3(1),   dim3(32),  0, stream,
                     ws, Wf, bfp, Wo, bo, out);
}
